// GatedDeltaNet_7945689498104
// MI455X (gfx1250) — matrix operand order repaired and checked by reading, not compiled, not run on hardware
//
#include <hip/hip_runtime.h>

typedef __attribute__((ext_vector_type(16))) _Float16 v16h;
typedef __attribute__((ext_vector_type(8)))  float    v8f;
typedef __attribute__((__vector_size__(16))) int      vi4;

#define L_SEQ   2048
#define H_HEADS 8
#define DK      128
#define DV      128
#define CIN     3072
#define KCONV   4
#define CHUNK   64
#define NCHUNK  (L_SEQ / CHUNK)

// padded LDS strides (bank-conflict-free fragment access)
#define LDQ   136   // f16 row stride for 64x128 chunk tensors (68 DW)
#define LDA   72    // f16 row stride for 64x64 T / A matrices (36 DW)
#define LDVT  72    // f16 row stride for transposed [128][64] tensors
#define LDST  132   // f32 row stride for transposed state St[dv][dk] (132 DW)

#ifndef __has_builtin
#define __has_builtin(x) 0
#endif

#if __has_builtin(__builtin_amdgcn_global_load_async_to_lds_b128)
#define HAVE_ASYNC_B128 1
#else
#define HAVE_ASYNC_B128 0
#endif
#if __has_builtin(__builtin_amdgcn_global_load_async_to_lds_b32)
#define HAVE_ASYNC_B32 1
#else
#define HAVE_ASYNC_B32 0
#endif

__device__ __forceinline__ void async_cp16(const void* g, void* l)
{
#if HAVE_ASYNC_B128
    __builtin_amdgcn_global_load_async_to_lds_b128(
        (__attribute__((address_space(1))) vi4*)g,
        (__attribute__((address_space(3))) vi4*)l, 0, 0);
#else
    *(uint4*)l = *(const uint4*)g;
#endif
}

__device__ __forceinline__ void async_cp4(const void* g, void* l)
{
#if HAVE_ASYNC_B32
    __builtin_amdgcn_global_load_async_to_lds_b32(
        (__attribute__((address_space(1))) int*)g,
        (__attribute__((address_space(3))) int*)l, 0, 0);
#else
    *(unsigned int*)l = *(const unsigned int*)g;
#endif
}

__device__ __forceinline__ void wait_async_zero()
{
#if __has_builtin(__builtin_amdgcn_s_wait_asynccnt)
    __builtin_amdgcn_s_wait_asynccnt(0);
#else
    asm volatile("s_wait_asynccnt 0" ::: "memory");
#endif
}

// ---------------------------------------------------------------------------
// WMMA fragment helpers (f16 A/B, f32 C/D); A/B use contiguous K order (same for both), not the ISA table order.
// A 16x32 (MxK): lane m<16 -> halves 0..15 = K0..15
//                lane m+16 -> halves 0..15 = K16..31
// B 32x16 (KxN): lane n<16 -> halves 0..15 = K0..15 ; lane n+16 -> K16..31
// C/D 16x16:     VGPR r -> row r (lanes 0-15) / r+8 (lanes 16-31), col = lane&15
// ---------------------------------------------------------------------------
__device__ __forceinline__ v16h ld_a16(const _Float16* p, int ld, int m0, int k0, int lane)
{
    const int m  = m0 + (lane & 15);
    const int kb = (lane & 16) ? 16 : 0;
    const _Float16* row = p + m * ld + k0;
    v16h a;
#pragma unroll
    for (int i = 0; i < 8; ++i) a[i]     = row[kb + i];
#pragma unroll
    for (int i = 0; i < 8; ++i) a[8 + i] = row[8 + kb + i];
    return a;
}

// A[m][k] = p[k][m] * scale[k]   (transposed source with per-K scaling: k^T)
__device__ __forceinline__ v16h ld_a16_T_scaleK(const _Float16* p, int ld, int m0, int k0,
                                                const float* scale, int lane)
{
    const int m  = m0 + (lane & 15);
    const int kb = (lane & 16) ? 16 : 0;
    v16h a;
#pragma unroll
    for (int i = 0; i < 8; ++i) {
        const int k = k0 + kb + i;
        a[i] = (_Float16)((float)p[k * ld + m] * scale[k]);
    }
#pragma unroll
    for (int i = 0; i < 8; ++i) {
        const int k = k0 + 8 + kb + i;
        a[8 + i] = (_Float16)((float)p[k * ld + m] * scale[k]);
    }
    return a;
}

// B[k][n] = p[n*ld + k]  (column-major / transposed source; contiguous per lane)
__device__ __forceinline__ v16h ld_b16_T(const _Float16* p, int ld, int k0, int n0, int lane)
{
    const int n  = n0 + (lane & 15);
    const int kb = k0 + ((lane & 16) ? 16 : 0);
    const _Float16* row = p + n * ld + kb;
    v16h b;
#pragma unroll
    for (int i = 0; i < 16; ++i) b[i] = row[i];
    return b;
}

// same, f32 source with convert (state St[dv][dk])
__device__ __forceinline__ v16h ld_b16_f32T(const float* p, int ld, int k0, int n0, int lane)
{
    const int n  = n0 + (lane & 15);
    const int kb = k0 + ((lane & 16) ? 16 : 0);
    const float* row = p + n * ld + kb;
    v16h b;
#pragma unroll
    for (int i = 0; i < 16; ++i) b[i] = (_Float16)row[i];
    return b;
}

__device__ __forceinline__ v8f wmma32(v16h a, v16h b, v8f c)
{
    return __builtin_amdgcn_wmma_f32_16x16x32_f16(false, a, false, b, (short)0, c, false, false);
}

#define V8F_ZERO {0.f,0.f,0.f,0.f,0.f,0.f,0.f,0.f}

// ---------------------------------------------------------------------------
// Kernel 1: causal depthwise conv + SiLU, split q/k/v, per-head L2 norm,
// gating g_raw = -exp(A_log)*softplus(a+dt_bias), beta = sigmoid(b).
// ---------------------------------------------------------------------------
__global__ __launch_bounds__(256) void gdn_conv_gate(
    const float* __restrict__ x, const float* __restrict__ av,
    const float* __restrict__ bv, const float* __restrict__ A_log,
    const float* __restrict__ dt_bias, const float* __restrict__ cw,
    const float* __restrict__ cb,
    _Float16* __restrict__ q16, _Float16* __restrict__ k16, _Float16* __restrict__ v16,
    float* __restrict__ graw, float* __restrict__ betab)
{
    __shared__ float y[CIN];
    __shared__ float nrm[16];
    const int l   = blockIdx.x;
    const int tid = threadIdx.x;
    if (tid < 16) nrm[tid] = 0.f;
    __syncthreads();

    for (int c = tid; c < CIN; c += 256) {
        float s = cb[c];
#pragma unroll
        for (int j = 0; j < KCONV; ++j) {
            const int t = l - (KCONV - 1) + j;
            const float xv = (t >= 0) ? x[(size_t)t * CIN + c] : 0.f;
            s += xv * cw[c * KCONV + j];
        }
        const float yy = s / (1.f + __expf(-s));   // SiLU
        y[c] = yy;
        if (c < 2048) atomicAdd(&nrm[c >> 7], yy * yy);
    }
    __syncthreads();

    for (int c = tid; c < CIN; c += 256) {
        const float yy = y[c];
        if (c < 1024) {
            const int hh = c >> 7, d = c & 127;
            const float r = rsqrtf(nrm[hh] + 1e-6f);
            q16[((size_t)hh * L_SEQ + l) * DK + d] = (_Float16)(yy * r * 0.08838834764831845f);
        } else if (c < 2048) {
            const int hh = (c - 1024) >> 7, d = c & 127;
            const float r = rsqrtf(nrm[8 + hh] + 1e-6f);
            k16[((size_t)hh * L_SEQ + l) * DK + d] = (_Float16)(yy * r);
        } else {
            const int hh = (c - 2048) >> 7, d = c & 127;
            v16[((size_t)hh * L_SEQ + l) * DV + d] = (_Float16)yy;
        }
    }
    if (tid < H_HEADS) {
        const float t  = av[(size_t)l * H_HEADS + tid] + dt_bias[tid];
        const float sp = (t > 20.f) ? t : log1pf(__expf(t));
        graw[(size_t)tid * L_SEQ + l]  = -__expf(A_log[tid]) * sp;
        betab[(size_t)tid * L_SEQ + l] = 1.f / (1.f + __expf(-bv[(size_t)l * H_HEADS + tid]));
    }
}

// ---------------------------------------------------------------------------
// Kernel 1b: inclusive within-chunk cumsum of g.
// ---------------------------------------------------------------------------
__global__ __launch_bounds__(64) void gdn_gscan(const float* __restrict__ graw,
                                                float* __restrict__ gcum)
{
    __shared__ float sg[CHUNK];
    const int h  = blockIdx.x >> 5;
    const int ic = blockIdx.x & 31;
    const int t  = threadIdx.x;
    sg[t] = graw[(size_t)h * L_SEQ + ic * CHUNK + t];
    __syncthreads();
    for (int off = 1; off < CHUNK; off <<= 1) {
        const float v = (t >= off) ? sg[t - off] : 0.f;
        __syncthreads();
        if (t >= off) sg[t] += v;
        __syncthreads();
    }
    gcum[(size_t)h * L_SEQ + ic * CHUNK + t] = sg[t];
}

// ---------------------------------------------------------------------------
// Kernel 2: WY representation per (head, chunk).
//   M[i][j] = -beta_i (k_i . k_j) exp(g_i - g_j)   (strict lower)
//   T = (I - M)^{-1}  (forward substitution)
//   w = T @ (k * beta * exp(g)),   u = T @ (v * beta)
// ---------------------------------------------------------------------------
__global__ __launch_bounds__(256) void gdn_wy(
    const _Float16* __restrict__ k16, const _Float16* __restrict__ v16,
    const float* __restrict__ gcum, const float* __restrict__ betab,
    _Float16* __restrict__ w16, _Float16* __restrict__ u16)
{
    __shared__ _Float16 sK[CHUNK * LDQ];
    __shared__ _Float16 sXt[DK * LDVT];        // X^T: [d][t]
    __shared__ float    sAf[CHUNK * CHUNK];
    __shared__ _Float16 sT[CHUNK * LDA];
    __shared__ float    sg[CHUNK], sb[CHUNK], rowbuf[CHUNK];

    const int h    = blockIdx.x >> 5;
    const int ic   = blockIdx.x & 31;
    const int base = ic * CHUNK;
    const int tid  = threadIdx.x;
    const int lane = tid & 31;
    const int wv   = tid >> 5;
    const int ro   = (lane & 16) ? 8 : 0;

    // async-stage k chunk into LDS while fetching gate scalars
    for (int i = tid; i < CHUNK * (DK / 8); i += 256) {
        const int t  = i >> 4;
        const int d8 = (i & 15) << 3;
        async_cp16(k16 + (size_t)(h * L_SEQ + base + t) * DK + d8, sK + t * LDQ + d8);
    }
    if (tid < CHUNK) {
        sg[tid] = gcum[(size_t)h * L_SEQ + base + tid];
        sb[tid] = betab[(size_t)h * L_SEQ + base + tid];
    }
    wait_async_zero();
    __syncthreads();

    // M = -(beta_i)(k_i . k_j)exp(g_i - g_j), strict lower.  16 tiles, 2/wave.
    for (int t = wv * 2; t < wv * 2 + 2; ++t) {
        const int tm = t >> 2, tn = t & 3;
        v8f acc = V8F_ZERO;
#pragma unroll
        for (int kk = 0; kk < 4; ++kk) {
            v16h a = ld_a16(sK, LDQ, tm * 16, kk * 32, lane);
            v16h b = ld_b16_T(sK, LDQ, kk * 32, tn * 16, lane);
            acc = wmma32(a, b, acc);
        }
#pragma unroll
        for (int r = 0; r < 8; ++r) {
            const int row = tm * 16 + r + ro;
            const int col = tn * 16 + (lane & 15);
            const float m = (row > col) ? -sb[row] * acc[r] * __expf(sg[row] - sg[col]) : 0.f;
            sAf[row * CHUNK + col] = m;
        }
    }
    __syncthreads();

    // forward substitution: T - I, rows ascending (reads original row via rowbuf)
    for (int i = 1; i < CHUNK; ++i) {
        if (tid < i) rowbuf[tid] = sAf[i * CHUNK + tid];
        __syncthreads();
        if (tid < i) {
            float s = rowbuf[tid];
            for (int t2 = tid + 1; t2 < i; ++t2)
                s += rowbuf[t2] * sAf[t2 * CHUNK + tid];
            sAf[i * CHUNK + tid] = s;
        }
        __syncthreads();
    }

    for (int i = tid; i < CHUNK * CHUNK; i += 256) {
        const int r = i >> 6, c = i & 63;
        const float tv = (r == c) ? 1.f : ((r > c) ? sAf[i] : 0.f);
        sT[r * LDA + c] = (_Float16)tv;
    }
    // X^T = (k * beta * exp(g))^T  -> sXt[d][t]
    for (int i = tid; i < CHUNK * DK; i += 256) {
        const int t = i >> 7, d = i & 127;
        sXt[d * LDVT + t] = (_Float16)((float)sK[t * LDQ + d] * sb[t] * __expf(sg[t]));
    }
    __syncthreads();

    // w = T @ X : [64,64]x[64,128], 32 tiles, 4/wave
    for (int t = wv; t < 32; t += 8) {
        const int tm = t >> 3, tn = t & 7;
        v8f acc = V8F_ZERO;
#pragma unroll
        for (int kk = 0; kk < 2; ++kk) {
            v16h a = ld_a16(sT, LDA, tm * 16, kk * 32, lane);
            v16h b = ld_b16_T(sXt, LDVT, kk * 32, tn * 16, lane);
            acc = wmma32(a, b, acc);
        }
#pragma unroll
        for (int r = 0; r < 8; ++r) {
            const int row = tm * 16 + r + ro;
            const int col = tn * 16 + (lane & 15);
            w16[(size_t)(h * L_SEQ + base + row) * DK + col] = (_Float16)acc[r];
        }
    }
    __syncthreads();

    // X^T = (v * beta)^T
    for (int i = tid; i < CHUNK * DV; i += 256) {
        const int t = i >> 7, d = i & 127;
        sXt[d * LDVT + t] =
            (_Float16)((float)v16[(size_t)(h * L_SEQ + base) * DV + i] * sb[t]);
    }
    __syncthreads();

    // u = T @ X
    for (int t = wv; t < 32; t += 8) {
        const int tm = t >> 3, tn = t & 7;
        v8f acc = V8F_ZERO;
#pragma unroll
        for (int kk = 0; kk < 2; ++kk) {
            v16h a = ld_a16(sT, LDA, tm * 16, kk * 32, lane);
            v16h b = ld_b16_T(sXt, LDVT, kk * 32, tn * 16, lane);
            acc = wmma32(a, b, acc);
        }
#pragma unroll
        for (int r = 0; r < 8; ++r) {
            const int row = tm * 16 + r + ro;
            const int col = tn * 16 + (lane & 15);
            u16[(size_t)(h * L_SEQ + base + row) * DV + col] = (_Float16)acc[r];
        }
    }
}

// ---------------------------------------------------------------------------
// Kernel 3: sequential inter-chunk scan, one block (8 waves) per head.
// Double-buffered async-to-LDS prefetch of the next chunk overlaps the WMMA
// work of the current chunk (ASYNCcnt / s_wait_asynccnt).
// Per chunk:  vnew = u - W@S ;  A = tril(Q@K^T . decay) ;
//             O = rowscale(Q@S, e^g) + A@vnew ;
//             S = e^{gC} S + (K . e^{gC-g})^T @ vnew
// State kept transposed: St[dv][dk] (f32, padded) for contiguous B fragments.
// ---------------------------------------------------------------------------
__global__ __launch_bounds__(256) void gdn_scan(
    const _Float16* __restrict__ q16, const _Float16* __restrict__ k16,
    const _Float16* __restrict__ w16, const _Float16* __restrict__ u16,
    const float* __restrict__ gcum, float* __restrict__ out)
{
    __shared__ float    St[DV * LDST];           // state, transposed [dv][dk]
    __shared__ _Float16 sQ[2][CHUNK * LDQ];
    __shared__ _Float16 sK[2][CHUNK * LDQ];
    __shared__ _Float16 sW[2][CHUNK * LDQ];
    __shared__ _Float16 sU[2][CHUNK * LDQ];
    __shared__ _Float16 sVt[DV * LDVT];          // vnew transposed [dv][t]
    __shared__ _Float16 sA[CHUNK * LDA];
    __shared__ float    sg[2][CHUNK];
    __shared__ float    ssc[CHUNK];              // exp(g_last - g_t)

    const int h    = blockIdx.x;
    const int tid  = threadIdx.x;
    const int lane = tid & 31;
    const int wv   = tid >> 5;
    const int ro   = (lane & 16) ? 8 : 0;

    const _Float16* qh = q16 + (size_t)h * L_SEQ * DK;
    const _Float16* kh = k16 + (size_t)h * L_SEQ * DK;
    const _Float16* wh = w16 + (size_t)h * L_SEQ * DK;
    const _Float16* uh = u16 + (size_t)h * L_SEQ * DV;
    const float*    gh = gcum + (size_t)h * L_SEQ;

    for (int i = tid; i < DV * LDST; i += 256) St[i] = 0.f;

    // prologue: stage chunk 0 into buffer 0
    {
        for (int i = tid; i < CHUNK * (DK / 8); i += 256) {
            const int t  = i >> 4;
            const int d8 = (i & 15) << 3;
            const size_t go = (size_t)t * DK + d8;
            const int    lo = t * LDQ + d8;
            async_cp16(qh + go, &sQ[0][lo]);
            async_cp16(kh + go, &sK[0][lo]);
            async_cp16(wh + go, &sW[0][lo]);
            async_cp16(uh + go, &sU[0][lo]);
        }
        if (tid < CHUNK) async_cp4(gh + tid, &sg[0][tid]);
    }
    wait_async_zero();
    __syncthreads();

    for (int ic = 0; ic < NCHUNK; ++ic) {
        const int cur  = ic & 1;
        const int base = ic * CHUNK;

        // prefetch next chunk into the other buffer (overlaps compute below)
        if (ic + 1 < NCHUNK) {
            const int nb = cur ^ 1;
            for (int i = tid; i < CHUNK * (DK / 8); i += 256) {
                const int t  = i >> 4;
                const int d8 = (i & 15) << 3;
                const size_t go = (size_t)(base + CHUNK + t) * DK + d8;
                const int    lo = t * LDQ + d8;
                async_cp16(qh + go, &sQ[nb][lo]);
                async_cp16(kh + go, &sK[nb][lo]);
                async_cp16(wh + go, &sW[nb][lo]);
                async_cp16(uh + go, &sU[nb][lo]);
            }
            if (tid < CHUNK) async_cp4(gh + base + CHUNK + tid, &sg[nb][tid]);
        }
        if (tid < CHUNK) ssc[tid] = __expf(sg[cur][CHUNK - 1] - sg[cur][tid]);

        // ---- step A: vnew = u - W @ S   (written transposed into sVt)
        for (int t = wv; t < 32; t += 8) {
            const int tm = t >> 3, tn = t & 7;
            v8f acc = V8F_ZERO;
#pragma unroll
            for (int kk = 0; kk < 4; ++kk) {
                v16h a = ld_a16(&sW[cur][0], LDQ, tm * 16, kk * 32, lane);
                v16h b = ld_b16_f32T(St, LDST, kk * 32, tn * 16, lane);
                acc = wmma32(a, b, acc);
            }
#pragma unroll
            for (int r = 0; r < 8; ++r) {
                const int row = tm * 16 + r + ro;
                const int col = tn * 16 + (lane & 15);
                const float vn = (float)sU[cur][row * LDQ + col] - acc[r];
                sVt[col * LDVT + row] = (_Float16)vn;
            }
        }

        // ---- step B: A = tril((Q @ K^T) * exp(g_i - g_j))
        for (int t = wv * 2; t < wv * 2 + 2; ++t) {
            const int tm = t >> 2, tn = t & 3;
            v8f acc = V8F_ZERO;
#pragma unroll
            for (int kk = 0; kk < 4; ++kk) {
                v16h a = ld_a16(&sQ[cur][0], LDQ, tm * 16, kk * 32, lane);
                v16h b = ld_b16_T(&sK[cur][0], LDQ, kk * 32, tn * 16, lane);
                acc = wmma32(a, b, acc);
            }
#pragma unroll
            for (int r = 0; r < 8; ++r) {
                const int row = tm * 16 + r + ro;
                const int col = tn * 16 + (lane & 15);
                const float v = (row >= col) ? acc[r] * __expf(sg[cur][row] - sg[cur][col]) : 0.f;
                sA[row * LDA + col] = (_Float16)v;
            }
        }
        __syncthreads();

        // ---- step C: O = rowscale(Q@S, e^g) + A @ vnew  -> global out
        for (int t = wv; t < 32; t += 8) {
            const int tm = t >> 3, tn = t & 7;
            v8f acc = V8F_ZERO;
#pragma unroll
            for (int kk = 0; kk < 4; ++kk) {
                v16h a = ld_a16(&sQ[cur][0], LDQ, tm * 16, kk * 32, lane);
                v16h b = ld_b16_f32T(St, LDST, kk * 32, tn * 16, lane);
                acc = wmma32(a, b, acc);
            }
#pragma unroll
            for (int r = 0; r < 8; ++r) {
                const int row = tm * 16 + r + ro;
                acc[r] *= __expf(sg[cur][row]);
            }
#pragma unroll
            for (int kk = 0; kk < 2; ++kk) {
                v16h a = ld_a16(sA, LDA, tm * 16, kk * 32, lane);
                v16h b = ld_b16_T(sVt, LDVT, kk * 32, tn * 16, lane);
                acc = wmma32(a, b, acc);
            }
#pragma unroll
            for (int r = 0; r < 8; ++r) {
                const int row = tm * 16 + r + ro;
                const int col = tn * 16 + (lane & 15);
                out[((size_t)(base + row) * H_HEADS + h) * DV + col] = acc[r];
            }
        }
        __syncthreads();

        // ---- step D: S = e^{gC} S + (K * e^{gC - g})^T @ vnew   (St in place)
        const float dec = __expf(sg[cur][CHUNK - 1]);
        for (int t = wv; t < 64; t += 8) {
            const int tm = t >> 3, tn = t & 7;   // tm: dk tile, tn: dv tile
            v8f acc;
#pragma unroll
            for (int r = 0; r < 8; ++r) {
                const int row = tm * 16 + r + ro;          // dk
                const int col = tn * 16 + (lane & 15);     // dv
                acc[r] = St[col * LDST + row] * dec;
            }
#pragma unroll
            for (int kk = 0; kk < 2; ++kk) {
                v16h a = ld_a16_T_scaleK(&sK[cur][0], LDQ, tm * 16, kk * 32, ssc, lane);
                v16h b = ld_b16_T(sVt, LDVT, kk * 32, tn * 16, lane);
                acc = wmma32(a, b, acc);
            }
#pragma unroll
            for (int r = 0; r < 8; ++r) {
                const int row = tm * 16 + r + ro;
                const int col = tn * 16 + (lane & 15);
                St[col * LDST + row] = acc[r];
            }
        }

        wait_async_zero();     // next chunk's async copies have landed
        __syncthreads();
    }
}

// ---------------------------------------------------------------------------
extern "C" void kernel_launch(void* const* d_in, const int* in_sizes, int n_in,
                              void* d_out, int out_size, void* d_ws, size_t ws_size,
                              hipStream_t stream)
{
    (void)in_sizes; (void)n_in; (void)out_size; (void)ws_size;
    const float* x       = (const float*)d_in[0];
    const float* a       = (const float*)d_in[1];
    const float* b       = (const float*)d_in[2];
    const float* A_log   = (const float*)d_in[3];
    const float* dt_bias = (const float*)d_in[4];
    const float* cw      = (const float*)d_in[5];
    const float* cb      = (const float*)d_in[6];
    float* out = (float*)d_out;

    const size_t nqkv = (size_t)H_HEADS * L_SEQ * DK;   // 2,097,152 elements
    _Float16* q16 = (_Float16*)d_ws;
    _Float16* k16 = q16 + nqkv;
    _Float16* v16 = k16 + nqkv;
    _Float16* w16 = v16 + nqkv;
    _Float16* u16 = w16 + nqkv;
    float* graw  = (float*)(u16 + nqkv);
    float* gcum  = graw + (size_t)H_HEADS * L_SEQ;
    float* betab = gcum + (size_t)H_HEADS * L_SEQ;

    gdn_conv_gate<<<L_SEQ, 256, 0, stream>>>(x, a, b, A_log, dt_bias, cw, cb,
                                             q16, k16, v16, graw, betab);
    gdn_gscan<<<H_HEADS * NCHUNK, 64, 0, stream>>>(graw, gcum);
    gdn_wy<<<H_HEADS * NCHUNK, 256, 0, stream>>>(k16, v16, gcum, betab, w16, u16);
    gdn_scan<<<H_HEADS, 256, 0, stream>>>(q16, k16, w16, u16, gcum, out);
}
